// BigramLanguageModel_84722524881037
// MI455X (gfx1250) — matrix operand order repaired and checked by reading, not compiled, not run on hardware
//
#include <hip/hip_runtime.h>
#include <hip/hip_bf16.h>

// ---------------------------------------------------------------------------
// Fused 6-layer GPT forward for MI455X (gfx1250, wave32, WMMA bf16).
// B=8192 T=8 V=65 C=512 H=8 HS=64 L=6 DFF=2048.
// Each workgroup (256 thr = 8 waves) owns 16 rows (= 2 batch elements) and
// runs the ENTIRE network with the activation tile resident in LDS (104 KB).
// Weights are pre-converted to bf16 [N][K] layout in d_ws (~38 MB, L2-hot).
// n-tiles are processed in PAIRS so one A fragment feeds two v_wmma ops.
// ---------------------------------------------------------------------------

#define B_    8192
#define T_    8
#define V_    65
#define C_    512
#define H_    8
#define HS_   64
#define L_    6
#define DFF_  2048
#define ROWS  16
#define LN_EPS 1e-5f
#define ATT_SCALE 0.044194173824159216f   // C^-0.5 (reference scales by C, not HS)

typedef __attribute__((ext_vector_type(16))) __bf16 v16bf;
typedef __attribute__((ext_vector_type(8)))  float  v8f;
typedef __attribute__((ext_vector_type(4)))  unsigned int u32x4;

union Frag { v16bf v; u32x4 q[2]; };

// Native bf16 convert (backend lowers fptrunc/fpext to gfx1250 bf16 cvt ops).
__device__ __forceinline__ unsigned short f2bf(float f) {
  __bf16 h = (__bf16)f;
  return __builtin_bit_cast(unsigned short, h);
}
__device__ __forceinline__ float bf2f(unsigned short u) {
  return (float)__builtin_bit_cast(__bf16, u);
}

// ---- single-tile K loop (used by LM head) ----------------------------------
__device__ __forceinline__ v8f wmma_kloop(const unsigned short* __restrict__ Arow,
                                          const unsigned short* __restrict__ Brow,
                                          int K, int kh, v8f acc) {
#pragma unroll 4
  for (int k0 = 0; k0 < K; k0 += 32) {
    if ((k0 & 96) == 0)
      __builtin_prefetch(Brow + k0 + 128, 0, 3);   // next 256B of this B row
    Frag a, b;
    a.q[0] = *(const u32x4*)(Arow + k0 + kh * 16);
    a.q[1] = *(const u32x4*)(Arow + k0 + 8 + kh * 16);
    b.q[0] = *(const u32x4*)(Brow + k0 + kh * 16);
    b.q[1] = *(const u32x4*)(Brow + k0 + 8 + kh * 16);
    acc = __builtin_amdgcn_wmma_f32_16x16x32_bf16(false, a.v, false, b.v,
                                                  (short)0, acc, false, false);
  }
  return acc;
}

// ---- paired-tile K loop: one A fragment -> two WMMAs -----------------------
__device__ __forceinline__ void wmma_kloop2(const unsigned short* __restrict__ Arow,
                                            const unsigned short* __restrict__ B0,
                                            const unsigned short* __restrict__ B1,
                                            int K, int kh, v8f& acc0, v8f& acc1) {
#pragma unroll 4
  for (int k0 = 0; k0 < K; k0 += 32) {
    if ((k0 & 96) == 0) {
      __builtin_prefetch(B0 + k0 + 128, 0, 3);
      __builtin_prefetch(B1 + k0 + 128, 0, 3);
    }
    Frag a, b0, b1;
    a.q[0]  = *(const u32x4*)(Arow + k0 + kh * 16);
    a.q[1]  = *(const u32x4*)(Arow + k0 + 8 + kh * 16);
    b0.q[0] = *(const u32x4*)(B0 + k0 + kh * 16);
    b0.q[1] = *(const u32x4*)(B0 + k0 + 8 + kh * 16);
    b1.q[0] = *(const u32x4*)(B1 + k0 + kh * 16);
    b1.q[1] = *(const u32x4*)(B1 + k0 + 8 + kh * 16);
    acc0 = __builtin_amdgcn_wmma_f32_16x16x32_bf16(false, a.v, false, b0.v,
                                                   (short)0, acc0, false, false);
    acc1 = __builtin_amdgcn_wmma_f32_16x16x32_bf16(false, a.v, false, b1.v,
                                                   (short)0, acc1, false, false);
  }
}

// LayerNorm over 16 rows x 512 cols; 16 threads/row, shuffle reduction inside
// 16-lane groups (wave32). Writes f32 residual + bf16 A-matrix copy.
__device__ __forceinline__ void layer_norm16(float* xres, unsigned short* xb,
                                             const float* g, const float* bb,
                                             int tid) {
  const int r = tid >> 4, c0 = tid & 15;
  float* row = xres + r * C_;
  float s = 0.f;
  for (int c = c0; c < C_; c += 16) s += row[c];
#pragma unroll
  for (int off = 8; off >= 1; off >>= 1) s += __shfl_xor(s, off, 32);
  const float mu = s * (1.f / C_);
  float vs = 0.f;
  for (int c = c0; c < C_; c += 16) { float d = row[c] - mu; vs += d * d; }
#pragma unroll
  for (int off = 8; off >= 1; off >>= 1) vs += __shfl_xor(vs, off, 32);
  const float rinv = rsqrtf(vs * (1.f / C_) + LN_EPS);
  for (int c = c0; c < C_; c += 16) {
    float v = (row[c] - mu) * rinv * g[c] + bb[c];
    row[c] = v;
    xb[r * C_ + c] = f2bf(v);
  }
}

struct Params {
  const int* idx;
  const float* tok; const float* pos;
  const float* proj_b; const float* b1; const float* b2;
  const float* ln1_g; const float* ln1_b; const float* ln2_g; const float* ln2_b;
  const float* lm_b;
  const unsigned short* wqkvT;   // [L][3][512(n)][512(k)] bf16
  const unsigned short* projT;   // [L][512(n)][512(k)]
  const unsigned short* w1T;     // [L][2048(n)][512(k)]
  const unsigned short* w2T;     // [L][512(n)][2048(k)]
  const unsigned short* lmT;     // [80(n, zero-padded)][512(k)]
  float* out;                    // [B*T][65] logits f32
};

__global__ __launch_bounds__(256) void gpt_fused(Params p) {
  extern __shared__ char smem[];
  float*          xres = (float*)smem;                       // 16x512 f32  32KB
  unsigned short* xb   = (unsigned short*)(smem + 32768);    // 16x512 bf16 16KB
  unsigned short* qkvs = (unsigned short*)(smem + 49152);    // 3x16x512    48KB
  unsigned short* hbuf = (unsigned short*)(smem + 98304);    // 16x256       8KB

  const int tid  = threadIdx.x;
  const int wave = tid >> 5;
  const int lane = tid & 31;
  const int ln16 = lane & 15;
  const int kh   = lane >> 4;
  const int row0 = blockIdx.x * ROWS;         // global row = b*T + t

  // ---- embedding: x = tok_table[idx] + pos_table ----
  for (int i = tid; i < ROWS * C_; i += 256) {
    int r = i >> 9, c = i & (C_ - 1);
    int grow = row0 + r;
    int b = grow >> 3, t = grow & 7;
    int tok = p.idx[b * T_ + t];
    float v = p.tok[(size_t)tok * C_ + c] + p.pos[t * C_ + c];
    xres[r * C_ + c] = v;
    xb[r * C_ + c]   = f2bf(v);
  }
  __syncthreads();

  const unsigned short* Arow0 = xb + ln16 * C_;   // this lane's A-matrix row

  for (int l = 0; l < L_; ++l) {
    // ================= QKV projection (M=16, N=1536, K=512) =================
    const unsigned short* Wqkv = p.wqkvT + (size_t)l * 3 * C_ * C_;
    for (int pq = 0; pq < 6; ++pq) {              // 96 n-tiles / 8 waves, paired
      int n0 = (wave * 12 + pq * 2) * 16;
      int n1 = n0 + 16;
      v8f acc0 = {0.f,0.f,0.f,0.f,0.f,0.f,0.f,0.f};
      v8f acc1 = acc0;
      wmma_kloop2(Arow0,
                  Wqkv + (size_t)(n0 + ln16) * C_,
                  Wqkv + (size_t)(n1 + ln16) * C_, C_, kh, acc0, acc1);
      // scatter both tiles into q/k/v LDS (pair never straddles a matrix)
      int mat = n0 >> 9;                           // 0=q 1=k 2=v
      unsigned short* d0 = qkvs + mat * (ROWS * C_) + (n0 & (C_ - 1)) + ln16;
      unsigned short* d1 = qkvs + mat * (ROWS * C_) + (n1 & (C_ - 1)) + ln16;
#pragma unroll
      for (int r = 0; r < 8; ++r) {
        d0[(r + 8 * kh) * C_] = f2bf(acc0[r]);
        d1[(r + 8 * kh) * C_] = f2bf(acc1[r]);
      }
    }
    __syncthreads();

    // ================= causal attention (T=8, per (b,h,t) thread) ===========
    if (tid < 128) {                               // waves 0-3 fully active
      int t = tid & 7, h = (tid >> 3) & 7, bl = tid >> 6;
      const unsigned short* qrow = qkvs + (bl * 8 + t) * C_ + h * HS_;
      float e[T_]; float mx = -1e30f;
#pragma unroll
      for (int s = 0; s < T_; ++s) {
        const unsigned short* krow = qkvs + ROWS * C_ + (bl * 8 + s) * C_ + h * HS_;
        float d = 0.f;
        for (int c = 0; c < HS_; ++c) d += bf2f(qrow[c]) * bf2f(krow[c]);
        e[s] = (s <= t) ? d * ATT_SCALE : -1e30f;
        mx = fmaxf(mx, e[s]);
      }
      float den = 0.f;
#pragma unroll
      for (int s = 0; s < T_; ++s) { e[s] = __expf(e[s] - mx); den += e[s]; }
      const float inv = 1.f / den;
      unsigned short* orow = xb + (bl * 8 + t) * C_ + h * HS_;  // o reuses xb
      for (int c = 0; c < HS_; ++c) {
        float o = 0.f;
#pragma unroll
        for (int s = 0; s < T_; ++s)
          o += e[s] * bf2f(qkvs[2 * ROWS * C_ + (bl * 8 + s) * C_ + h * HS_ + c]);
        orow[c] = f2bf(o * inv);
      }
    }
    __syncthreads();

    // ================= output projection + residual =========================
    const unsigned short* Wp = p.projT + (size_t)l * C_ * C_;
    const float* pb = p.proj_b + l * C_;
    for (int pq = 0; pq < 2; ++pq) {              // 32 n-tiles / 8 waves, paired
      int n0 = (wave * 4 + pq * 2) * 16;
      int n1 = n0 + 16;
      float bi0 = pb[n0 + ln16], bi1 = pb[n1 + ln16];
      v8f acc0 = {bi0,bi0,bi0,bi0,bi0,bi0,bi0,bi0};
      v8f acc1 = {bi1,bi1,bi1,bi1,bi1,bi1,bi1,bi1};
      wmma_kloop2(Arow0,
                  Wp + (size_t)(n0 + ln16) * C_,
                  Wp + (size_t)(n1 + ln16) * C_, C_, kh, acc0, acc1);
      float* r0 = xres + n0 + ln16;
      float* r1 = xres + n1 + ln16;
#pragma unroll
      for (int r = 0; r < 8; ++r) {
        r0[(r + 8 * kh) * C_] += acc0[r];
        r1[(r + 8 * kh) * C_] += acc1[r];
      }
    }
    __syncthreads();
    layer_norm16(xres, xb, p.ln1_g + l * C_, p.ln1_b + l * C_, tid);
    __syncthreads();

    // ================= MLP: y += relu(x@W1+b1)@W2+b2, DFF chunked ===========
    const unsigned short* W1 = p.w1T + (size_t)l * DFF_ * C_;
    const unsigned short* W2 = p.w2T + (size_t)l * C_ * DFF_;
    const float* b1p = p.b1 + l * DFF_;
    const float* b2p = p.b2 + l * C_;
    v8f acc2[4];
#pragma unroll
    for (int j = 0; j < 4; ++j) {
      float b = b2p[(wave * 4 + j) * 16 + ln16];
      acc2[j] = {b,b,b,b,b,b,b,b};
    }
    for (int chunk = 0; chunk < DFF_ / 256; ++chunk) {
      int f0 = chunk * 256;
      {                                            // 16 h-tiles / 8 waves, paired
        int nf0 = f0 + (wave * 2) * 16;
        int nf1 = nf0 + 16;
        float b0 = b1p[nf0 + ln16], b1v = b1p[nf1 + ln16];
        v8f h0 = {b0,b0,b0,b0,b0,b0,b0,b0};
        v8f h1 = {b1v,b1v,b1v,b1v,b1v,b1v,b1v,b1v};
        wmma_kloop2(Arow0,
                    W1 + (size_t)(nf0 + ln16) * C_,
                    W1 + (size_t)(nf1 + ln16) * C_, C_, kh, h0, h1);
        unsigned short* d0 = hbuf + (nf0 - f0) + ln16;
        unsigned short* d1 = hbuf + (nf1 - f0) + ln16;
#pragma unroll
        for (int r = 0; r < 8; ++r) {
          d0[(r + 8 * kh) * 256] = f2bf(fmaxf(h0[r], 0.f));   // ReLU
          d1[(r + 8 * kh) * 256] = f2bf(fmaxf(h1[r], 0.f));
        }
      }
      __syncthreads();
      const unsigned short* Hrow = hbuf + ln16 * 256;
#pragma unroll
      for (int j = 0; j < 2; ++j) {
        int n0 = (wave * 4 + j * 2) * 16;
        int n1 = n0 + 16;
        wmma_kloop2(Hrow,
                    W2 + (size_t)(n0 + ln16) * DFF_ + f0,
                    W2 + (size_t)(n1 + ln16) * DFF_ + f0,
                    256, kh, acc2[j * 2], acc2[j * 2 + 1]);
      }
      __syncthreads();
    }
    for (int j = 0; j < 4; ++j) {
      int n0 = (wave * 4 + j) * 16;
      float* dst = xres + n0 + ln16;
#pragma unroll
      for (int r = 0; r < 8; ++r) dst[(r + 8 * kh) * C_] += acc2[j][r];
    }
    __syncthreads();
    layer_norm16(xres, xb, p.ln2_g + l * C_, p.ln2_b + l * C_, tid);
    __syncthreads();
  }

  // ================= LM head (N=65 padded to 80 -> 5 tiles) =================
  if (wave < 5) {
    int n0 = wave * 16;
    int n  = n0 + ln16;
    float bias = (n < V_) ? p.lm_b[n] : 0.f;
    v8f acc = {bias,bias,bias,bias,bias,bias,bias,bias};
    acc = wmma_kloop(Arow0, p.lmT + (size_t)n * C_, C_, kh, acc);
    if (n < V_) {
#pragma unroll
      for (int r = 0; r < 8; ++r) {
        int grow = row0 + r + 8 * kh;
        p.out[(size_t)grow * V_ + n] = acc[r];
      }
    }
  }
}

// ---------------------------------------------------------------------------
// Weight prep: f32 -> bf16, transposed to [N][K] so WMMA B-fragments are two
// contiguous 16B loads per lane.
// ---------------------------------------------------------------------------
__global__ void prep_qkv(const float* wq, const float* wk, const float* wv,
                         unsigned short* out, int total) {
  int i = blockIdx.x * 256 + threadIdx.x;
  if (i >= total) return;
  int k = i & 511;
  int rest = i >> 9;
  int n = rest & 511; rest >>= 9;
  int mat = rest % 3, l = rest / 3;
  int h = n >> 6, d = n & 63;                      // n = h*HS + d
  const float* src = (mat == 0) ? wq : (mat == 1) ? wk : wv;
  out[i] = f2bf(src[(((size_t)l * H_ + h) * C_ + k) * HS_ + d]);
}

__global__ void prep_transpose(const float* src, unsigned short* out,
                               int K, int N, int total) {
  int i = blockIdx.x * 256 + threadIdx.x;
  if (i >= total) return;
  int k = i % K;
  int rest = i / K;
  int n = rest % N;
  int l = rest / N;
  out[i] = f2bf(src[((size_t)l * K + k) * N + n]);
}

__global__ void prep_lm(const float* lm_w, unsigned short* out) {
  int i = blockIdx.x * 256 + threadIdx.x;
  if (i >= 80 * C_) return;
  int k = i & 511, n = i >> 9;
  out[i] = (n < V_) ? f2bf(lm_w[(size_t)k * V_ + n]) : (unsigned short)0;
}

// ---------------------------------------------------------------------------
extern "C" void kernel_launch(void* const* d_in, const int* in_sizes, int n_in,
                              void* d_out, int out_size, void* d_ws, size_t ws_size,
                              hipStream_t stream) {
  const int*   idx    = (const int*)  d_in[0];
  const float* tok    = (const float*)d_in[1];
  const float* pos    = (const float*)d_in[2];
  const float* wq     = (const float*)d_in[3];
  const float* wk     = (const float*)d_in[4];
  const float* wv     = (const float*)d_in[5];
  const float* proj_w = (const float*)d_in[6];
  const float* proj_b = (const float*)d_in[7];
  const float* w1     = (const float*)d_in[8];
  const float* b1     = (const float*)d_in[9];
  const float* w2     = (const float*)d_in[10];
  const float* b2     = (const float*)d_in[11];
  const float* ln1_g  = (const float*)d_in[12];
  const float* ln1_b  = (const float*)d_in[13];
  const float* ln2_g  = (const float*)d_in[14];
  const float* ln2_b  = (const float*)d_in[15];
  const float* lm_w   = (const float*)d_in[16];
  const float* lm_b   = (const float*)d_in[17];

  // bf16 workspace layout (ushort elements)
  unsigned short* ws    = (unsigned short*)d_ws;
  unsigned short* wqkvT = ws;                      // 6*3*512*512 = 4,718,592
  unsigned short* projT = ws + 4718592;            // 6*512*512   = 1,572,864
  unsigned short* w1T   = ws + 6291456;            // 6*2048*512  = 6,291,456
  unsigned short* w2T   = ws + 12582912;           // 6*512*2048  = 6,291,456
  unsigned short* lmT   = ws + 18874368;           // 80*512      = 40,960
  // total ~37.8 MB of d_ws

  {
    int tot = L_ * 3 * C_ * C_;
    prep_qkv<<<(tot + 255) / 256, 256, 0, stream>>>(wq, wk, wv, wqkvT, tot);
  }
  {
    int tot = L_ * C_ * C_;
    prep_transpose<<<(tot + 255) / 256, 256, 0, stream>>>(proj_w, projT, C_, C_, tot);
  }
  {
    int tot = L_ * DFF_ * C_;   // out elems: [L][N=2048][K=512]
    prep_transpose<<<(tot + 255) / 256, 256, 0, stream>>>(w1, w1T, C_, DFF_, tot);
  }
  {
    int tot = L_ * C_ * DFF_;   // out elems: [L][N=512][K=2048]
    prep_transpose<<<(tot + 255) / 256, 256, 0, stream>>>(w2, w2T, DFF_, C_, tot);
  }
  prep_lm<<<(80 * C_ + 255) / 256, 256, 0, stream>>>(lm_w, lmT);

  Params p;
  p.idx = idx; p.tok = tok; p.pos = pos;
  p.proj_b = proj_b; p.b1 = b1; p.b2 = b2;
  p.ln1_g = ln1_g; p.ln1_b = ln1_b; p.ln2_g = ln2_g; p.ln2_b = ln2_b;
  p.lm_b = lm_b;
  p.wqkvT = wqkvT; p.projT = projT; p.w1T = w1T; p.w2T = w2T; p.lmT = lmT;
  p.out = (float*)d_out;

  const int    grid  = (B_ * T_) / ROWS;           // 4096 workgroups
  const size_t shmem = 32768 + 16384 + 49152 + 8192;   // 104 KB dynamic LDS
  gpt_fused<<<grid, 256, shmem, stream>>>(p);
}
